// MOGEEGModel_55456617726336
// MI455X (gfx1250) — matrix operand order repaired and checked by reading, not compiled, not run on hardware
//
#include <hip/hip_runtime.h>

typedef __attribute__((ext_vector_type(16))) _Float16 v16h;
typedef __attribute__((ext_vector_type(8)))  _Float16 v8h;
typedef __attribute__((ext_vector_type(8)))  float    v8f;

#define DEVI __device__ __forceinline__

// ---------------- math helpers ----------------
DEVI float fast_tanh(float x) {
#if __has_builtin(__builtin_amdgcn_tanhf)
  return __builtin_amdgcn_tanhf(x);
#elif __has_builtin(__builtin_amdgcn_tanh_f32)
  return __builtin_amdgcn_tanh_f32(x);
#else
  float e = __expf(-2.0f * x);
  return (1.0f - e) * __builtin_amdgcn_rcpf(1.0f + e);
#endif
}
DEVI float sig2(float x) { return 1.0f + fast_tanh(0.5f * x); }        // 2*sigmoid(x)
DEVI float sig1(float x) { return 0.5f + 0.5f * fast_tanh(0.5f * x); } // sigmoid(x)

// butterfly sum over the 16 lanes of each half-wave (column lanes)
DEVI float rsum16(float x) {
  x += __shfl_xor(x, 1, 32);
  x += __shfl_xor(x, 2, 32);
  x += __shfl_xor(x, 4, 32);
  x += __shfl_xor(x, 8, 32);
  return x;
}

DEVI v8f wmma_hf(v16h a, v16h b, v8f c) {
  return __builtin_amdgcn_wmma_f32_16x16x32_f16(false, a, false, b, (short)0, c,
                                                false, false);
}

DEVI v16h pack16(float4 a, float4 b, float4 c, float4 d) {
  v16h r;
  r[0]=(_Float16)a.x;  r[1]=(_Float16)a.y;  r[2]=(_Float16)a.z;  r[3]=(_Float16)a.w;
  r[4]=(_Float16)b.x;  r[5]=(_Float16)b.y;  r[6]=(_Float16)b.z;  r[7]=(_Float16)b.w;
  r[8]=(_Float16)c.x;  r[9]=(_Float16)c.y;  r[10]=(_Float16)c.z; r[11]=(_Float16)c.w;
  r[12]=(_Float16)d.x; r[13]=(_Float16)d.y; r[14]=(_Float16)d.z; r[15]=(_Float16)d.w;
  return r;
}

// Load a 16x32 f16 A-fragment (one K-half of a 16x64 row-major tile) from LDS.
// Lane layout: row = lane%16; halves 0..7 = K[hi*8 .. +7], halves 8..15 = K[16+hi*8 .. +7].
DEVI v16h load_afrag(const _Float16* hrow, int f, int hi) {
  v8h lo = *(const v8h*)(hrow + f * 32 + hi * 8);
  v8h hh = *(const v8h*)(hrow + f * 32 + 16 + hi * 8);
  return __builtin_shufflevector(lo, hh, 0, 1, 2, 3, 4, 5, 6, 7, 8, 9, 10, 11, 12,
                                 13, 14, 15);
}

// ---------------- weight prep: f32 row-major -> f16 WMMA B-fragment image ----------------
// B-fragment (32K x 16N) element (k,n): lane = (n%16) + 16*((k%16)/8), half = (k%8) + 8*((k%32)/16).
// stemF: 4 layers * 8 frags * 512 halves ; whhF: 32 frags * 512 halves.
__global__ __launch_bounds__(256) void prep_weights(
    const float* __restrict__ W1, const float* __restrict__ W2,
    const float* __restrict__ W3, const float* __restrict__ W4,
    const float* __restrict__ Whh, _Float16* __restrict__ stemF,
    _Float16* __restrict__ whhF) {
  int id = blockIdx.x * 256 + threadIdx.x;
  if (id >= 16384) return;
  int lane = (id >> 4) & 31;
  int j = id & 15;
  // stem fragment
  {
    int L = id >> 12;
    int rem = id & 4095;
    int fi = rem >> 9;              // frag within layer: nt*2+kf
    int nt = fi >> 1, kf = fi & 1;
    int n = nt * 16 + (lane & 15);
    int k = kf * 32 + (lane >> 4) * 8 + (j & 7) + (j >> 3) * 16;
    const float* W = (L == 0) ? W1 : (L == 1) ? W2 : (L == 2) ? W3 : W4;
    stemF[id] = (_Float16)W[n * 64 + k];   // B[k][n] = W[n][k]
  }
  // Whh fragment
  {
    int fi = id >> 9;               // 0..31 : nt*2+kf, nt over 16 gate-col tiles
    int n = (fi >> 1) * 16 + (lane & 15);
    int k = (fi & 1) * 32 + (lane >> 4) * 8 + (j & 7) + (j >> 3) * 16;
    whhF[id] = (_Float16)Whh[n * 64 + k];
  }
}

// ---------------- fused mogrifier-LSTM kernel ----------------
// block = 128 threads = 4 waves; each wave owns a 16-row batch tile (64 rows/block).
__global__ __launch_bounds__(128) void mog_lstm_kernel(
    const float* __restrict__ x, const float* __restrict__ B1,
    const float* __restrict__ B2, const float* __restrict__ B3,
    const float* __restrict__ B4, const float* __restrict__ M0w,
    const float* __restrict__ M0b, const float* __restrict__ M1w,
    const float* __restrict__ M1b, const float* __restrict__ M2w,
    const float* __restrict__ M2b, const float* __restrict__ M3w,
    const float* __restrict__ M3b, const float* __restrict__ M4w,
    const float* __restrict__ M4b, const float* __restrict__ Wih,
    const float* __restrict__ Bih, const float* __restrict__ Bhh,
    const float* __restrict__ Wp, const float* __restrict__ Bp,
    const _Float16* __restrict__ stemF, const _Float16* __restrict__ whhF,
    float* __restrict__ out, int Bn) {
  __shared__ uint4 sWhhRaw[2048];                 // 32 KB: Whh f16 B-fragments
  __shared__ __align__(16) _Float16 sH[4][1024];  // 8 KB: per-wave h staging (16x64)
  __shared__ __align__(16) _Float16 sX[4][1024];  // 8 KB: per-wave stem output (16x64)

  const int tid = threadIdx.x;
  const int wv = tid >> 5;
  const int lane = tid & 31;
  const int li = lane & 15;  // column lane
  const int hi = lane >> 4;  // row-half select

  // cooperative copy of Whh fragments into LDS (shared by all 4 waves)
  {
    const uint4* src = (const uint4*)whhF;
    for (int i = tid; i < 2048; i += 128) sWhhRaw[i] = src[i];
  }
  __syncthreads();
  const v16h* wB = (const v16h*)sWhhRaw;

  const int row0 = blockIdx.x * 64 + wv * 16;
  if (row0 >= Bn) return;
  _Float16* sHw = sH[wv];
  _Float16* sXw = sX[wv];

  // per-lane mogrifier constants (this lane's 4 columns: nt*16+li)
  float m0w[4], m1w[4], m1b[4], m2w[4], m3w[4], m3b[4], m4w[4];
#pragma unroll
  for (int nt = 0; nt < 4; ++nt) {
    int c = nt * 16 + li;
    m0w[nt] = M0w[c]; m1w[nt] = M1w[c]; m1b[nt] = M1b[c];
    m2w[nt] = M2w[c]; m3w[nt] = M3w[c]; m3b[nt] = M3b[c];
    m4w[nt] = M4w[c];
  }
  const float m0b = M0b[0], m2b = M2b[0], m4b = M4b[0];

  // per-lane LSTM input weights / fused biases, column-indexed (gate g, H-chunk jH)
  float wih[16], gb[16];
#pragma unroll
  for (int g = 0; g < 4; ++g)
#pragma unroll
    for (int jH = 0; jH < 4; ++jH) {
      int n = g * 64 + jH * 16 + li;
      wih[g * 4 + jH] = Wih[n];
      gb[g * 4 + jH] = Bih[n] + Bhh[n];
    }

  // ---- load x tile directly as f16 A-fragments ----
  v16h a0, a1;
  {
    const float* xr = x + (size_t)(row0 + li) * 64;
#pragma unroll
    for (int f = 0; f < 2; ++f) {
      int b = f * 32 + hi * 8;
      float4 p0 = *(const float4*)(xr + b);
      float4 p1 = *(const float4*)(xr + b + 4);
      float4 q0 = *(const float4*)(xr + b + 16);
      float4 q1 = *(const float4*)(xr + b + 20);
      v16h a = pack16(p0, p1, q0, q1);
      if (f == 0) a0 = a; else a1 = a;
    }
  }

  // ---- MLP stem: 4x (Linear(64,64)+ReLU) via WMMA ----
  float y[4][8];
  const v16h* sf = (const v16h*)stemF;
#pragma unroll
  for (int L = 0; L < 4; ++L) {
    v8f acc[4];
#pragma unroll
    for (int nt = 0; nt < 4; ++nt) {
      v8f z = {};
      z = wmma_hf(a0, sf[(L * 8 + nt * 2 + 0) * 32 + lane], z);
      z = wmma_hf(a1, sf[(L * 8 + nt * 2 + 1) * 32 + lane], z);
      acc[nt] = z;
    }
    const float* BL = (L == 0) ? B1 : (L == 1) ? B2 : (L == 2) ? B3 : B4;
#pragma unroll
    for (int nt = 0; nt < 4; ++nt) {
      float bb = BL[nt * 16 + li];
#pragma unroll
      for (int v = 0; v < 8; ++v) {
        float t = acc[nt][v] + bb;
        y[nt][v] = t > 0.f ? t : 0.f;
      }
    }
    if (L < 3) {  // stage through LDS to rebuild next A-fragments
#pragma unroll
      for (int nt = 0; nt < 4; ++nt)
#pragma unroll
        for (int v = 0; v < 8; ++v)
          sHw[(v + 8 * hi) * 64 + nt * 16 + li] = (_Float16)y[nt][v];
      asm volatile("s_wait_dscnt 0x0" ::: "memory");
      a0 = load_afrag(sHw + li * 64, 0, hi);
      a1 = load_afrag(sHw + li * 64, 1, hi);
    }
  }
  // stash stem output (timestep scalar inputs), row-major f16
#pragma unroll
  for (int nt = 0; nt < 4; ++nt)
#pragma unroll
    for (int v = 0; v < 8; ++v)
      sXw[(v + 8 * hi) * 64 + nt * 16 + li] = (_Float16)y[nt][v];
  asm volatile("s_wait_dscnt 0x0" ::: "memory");

  // ---- recurrence ----
  float h[4][8] = {}, c[4][8] = {};
  for (int t = 0; t < 64; ++t) {
    float xt0[8], xt1[8], xt2[8], xtf[8], p[8];
#pragma unroll
    for (int v = 0; v < 8; ++v) xt0[v] = (float)sXw[(v + 8 * hi) * 64 + t];

    // mogrify step 0: a0 = 2*sig(h.M0w + b); xt1 = a0*xt0
#pragma unroll
    for (int v = 0; v < 8; ++v)
      p[v] = h[0][v] * m0w[0] + h[1][v] * m0w[1] + h[2][v] * m0w[2] + h[3][v] * m0w[3];
#pragma unroll
    for (int v = 0; v < 8; ++v) xt1[v] = sig2(rsum16(p[v]) + m0b) * xt0[v];

    // step 1+2: vh = 2*sig(xt1*M1w+M1b)*h ; d2 = vh.M2w ; xt2 = 2*sig(d2+b)*xt1
    float vh[4][8];
#pragma unroll
    for (int v = 0; v < 8; ++v) p[v] = 0.f;
#pragma unroll
    for (int nt = 0; nt < 4; ++nt)
#pragma unroll
      for (int v = 0; v < 8; ++v) {
        float v1 = sig2(xt1[v] * m1w[nt] + m1b[nt]);
        vh[nt][v] = v1 * h[nt][v];
        p[v] += vh[nt][v] * m2w[nt];
      }
#pragma unroll
    for (int v = 0; v < 8; ++v) xt2[v] = sig2(rsum16(p[v]) + m2b) * xt1[v];

    // step 3+4: h_final = 2*sig(xt2*M3w+M3b)*vh ; d4 = h_final.M4w ; xtf = 2*sig(d4+b)*xt2
#pragma unroll
    for (int v = 0; v < 8; ++v) p[v] = 0.f;
#pragma unroll
    for (int nt = 0; nt < 4; ++nt)
#pragma unroll
      for (int v = 0; v < 8; ++v) {
        float v3 = sig2(xt2[v] * m3w[nt] + m3b[nt]);
        float hf = v3 * vh[nt][v];
        h[nt][v] = hf;
        p[v] += hf * m4w[nt];
      }
#pragma unroll
    for (int v = 0; v < 8; ++v) xtf[v] = sig2(rsum16(p[v]) + m4b) * xt2[v];

    // stage mogrified h as f16, rebuild A-fragments
#pragma unroll
    for (int nt = 0; nt < 4; ++nt)
#pragma unroll
      for (int v = 0; v < 8; ++v)
        sHw[(v + 8 * hi) * 64 + nt * 16 + li] = (_Float16)h[nt][v];
    asm volatile("s_wait_dscnt 0x0" ::: "memory");
    v16h A0 = load_afrag(sHw + li * 64, 0, hi);
    v16h A1 = load_afrag(sHw + li * 64, 1, hi);

    // gates = h @ Whh.T (WMMA) + xt*Wih + Bih + Bhh ; LSTM update
#pragma unroll
    for (int jH = 0; jH < 4; ++jH) {
      v8f g4[4];
#pragma unroll
      for (int g = 0; g < 4; ++g) {
        int ntg = g * 4 + jH;
        v8f z = {};
        z = wmma_hf(A0, wB[(ntg * 2 + 0) * 32 + lane], z);
        z = wmma_hf(A1, wB[(ntg * 2 + 1) * 32 + lane], z);
        g4[g] = z;
      }
#pragma unroll
      for (int v = 0; v < 8; ++v) {
        float iv = g4[0][v] + xtf[v] * wih[0 + jH] + gb[0 + jH];
        float fv = g4[1][v] + xtf[v] * wih[4 + jH] + gb[4 + jH];
        float gv = g4[2][v] + xtf[v] * wih[8 + jH] + gb[8 + jH];
        float ov = g4[3][v] + xtf[v] * wih[12 + jH] + gb[12 + jH];
        float cn = sig1(fv) * c[jH][v] + sig1(iv) * fast_tanh(gv);
        c[jH][v] = cn;
        h[jH][v] = sig1(ov) * fast_tanh(cn);
      }
    }
  }

  // ---- final projection hT @ Wp.T + Bp ----
#pragma unroll
  for (int nt = 0; nt < 4; ++nt)
#pragma unroll
    for (int v = 0; v < 8; ++v)
      sHw[(v + 8 * hi) * 64 + nt * 16 + li] = (_Float16)h[nt][v];
  asm volatile("s_wait_dscnt 0x0" ::: "memory");
  if (hi == 0) {  // one lane per row
    const _Float16* hr = sHw + li * 64;
    float o[5];
#pragma unroll
    for (int q = 0; q < 5; ++q) o[q] = Bp[q];
    for (int cix = 0; cix < 64; ++cix) {
      float hv = (float)hr[cix];
#pragma unroll
      for (int q = 0; q < 5; ++q) o[q] += hv * Wp[q * 64 + cix];
    }
    float* op = out + (size_t)(row0 + li) * 5;
#pragma unroll
    for (int q = 0; q < 5; ++q) op[q] = o[q];
  }
}

// ---------------- launch ----------------
extern "C" void kernel_launch(void* const* d_in, const int* in_sizes, int n_in,
                              void* d_out, int out_size, void* d_ws, size_t ws_size,
                              hipStream_t stream) {
  const float* x   = (const float*)d_in[0];
  const float* W1  = (const float*)d_in[1];
  const float* B1  = (const float*)d_in[2];
  const float* W2  = (const float*)d_in[3];
  const float* B2  = (const float*)d_in[4];
  const float* W3  = (const float*)d_in[5];
  const float* B3  = (const float*)d_in[6];
  const float* W4  = (const float*)d_in[7];
  const float* B4  = (const float*)d_in[8];
  const float* M0w = (const float*)d_in[9];
  const float* M0b = (const float*)d_in[10];
  const float* M1w = (const float*)d_in[11];
  const float* M1b = (const float*)d_in[12];
  const float* M2w = (const float*)d_in[13];
  const float* M2b = (const float*)d_in[14];
  const float* M3w = (const float*)d_in[15];
  const float* M3b = (const float*)d_in[16];
  const float* M4w = (const float*)d_in[17];
  const float* M4b = (const float*)d_in[18];
  const float* Wih = (const float*)d_in[19];
  const float* Bih = (const float*)d_in[20];
  const float* Whh = (const float*)d_in[21];
  const float* Bhh = (const float*)d_in[22];
  const float* Wp  = (const float*)d_in[23];
  const float* Bp  = (const float*)d_in[24];

  _Float16* stemF = (_Float16*)d_ws;       // 16384 halves
  _Float16* whhF  = stemF + 16384;         // 16384 halves

  prep_weights<<<64, 256, 0, stream>>>(W1, W2, W3, W4, Whh, stemF, whhF);

  int Bn = in_sizes[0] / 64;
  int grid = (Bn + 63) / 64;
  mog_lstm_kernel<<<grid, 128, 0, stream>>>(
      x, B1, B2, B3, B4, M0w, M0b, M1w, M1b, M2w, M2b, M3w, M3b, M4w, M4b, Wih,
      Bih, Bhh, Wp, Bp, stemF, whhF, (float*)d_out, Bn);
}
